// Generator_18056042512602
// MI455X (gfx1250) — matrix operand order repaired and checked by reading, not compiled, not run on hardware
//
#include <hip/hip_runtime.h>
#include <hip/hip_bf16.h>
#include <stdint.h>

// ---------------------------------------------------------------------------
// 2-layer GCN for MI455X (gfx1250, wave32).
//   GEMMs via v_wmma_f32_16x16x32_bf16 (bf16 A/B, f32 accum).
//   Edge scatter: bf16 gather (L2-resident feature matrix) + f32 atomics.
//   Self-loops folded into epilogues: out[i] += h[i]/deg[i] + b.
// ---------------------------------------------------------------------------

typedef __attribute__((ext_vector_type(16))) __bf16 v16bf;
typedef __attribute__((ext_vector_type(8)))  float  v8f;

union PackedHalf2 { __bf16 h[2]; uint32_t u; };
union BlockHalf   { uint4 u[2]; v16bf v; };

// ---------------- degree / normalization ----------------------------------

__global__ void k_init_deg(float* __restrict__ deg, int n) {
  int i = blockIdx.x * blockDim.x + threadIdx.x;
  if (i < n) deg[i] = 1.0f;                      // self-loop contributes 1
}

__global__ void k_accum_deg(const long long* __restrict__ dst,
                            float* __restrict__ deg, int e) {
  int i = blockIdx.x * blockDim.x + threadIdx.x;
  if (i < e) atomicAdd(&deg[(int)dst[i]], 1.0f);
}

__global__ void k_finalize_dinv(float* __restrict__ deg, int n) {
  int i = blockIdx.x * blockDim.x + threadIdx.x;
  if (i < n) deg[i] = rsqrtf(deg[i]);            // deg >= 1 always
}

__global__ void k_zero_f32(float* __restrict__ p, long long n) {
  long long i = (long long)blockIdx.x * blockDim.x + threadIdx.x;
  if (i < n) p[i] = 0.0f;
}

// ---------------- bf16 packing into WMMA register order --------------------
// A (16x32 MxK blocks): lane L (m = L&15, h = L>>4), dword i (0..7):
//   K = ((i>>2)<<4) + h*8 + ((i&3)<<1)          (ISA 7.12.2, 16-bit A 16x32)
__global__ void k_pack_a(const float* __restrict__ X, uint32_t* __restrict__ out,
                         int M, int K) {
  int t = blockIdx.x * blockDim.x + threadIdx.x;
  int KT = K >> 5;
  int total = (M >> 4) * KT * 256;
  if (t >= total) return;
  int block = t >> 8, r = t & 255;
  int lane = r >> 3, i = r & 7;
  int bm = block / KT, bk = block % KT;
  int m = bm * 16 + (lane & 15);
  int k = bk * 32 + ((i >> 2) << 4) + ((lane >> 4) << 3) + ((i & 3) << 1);
  const float* p = X + (size_t)m * K + k;
  PackedHalf2 ph;
  ph.h[0] = (__bf16)p[0];
  ph.h[1] = (__bf16)p[1];
  out[t] = ph.u;
}

// B (32x16 KxN blocks): lane L (n = L&15, h = L>>4), dword i (0..7):
//   K = ((i>>2)<<4) + h*8 + ((i&3)<<1)          (ISA 7.12.2/7.12.4 B layout)
__global__ void k_pack_b(const float* __restrict__ W, uint32_t* __restrict__ out,
                         int K, int N) {
  int t = blockIdx.x * blockDim.x + threadIdx.x;
  int NT = N >> 4;
  int total = (K >> 5) * NT * 256;
  if (t >= total) return;
  int block = t >> 8, r = t & 255;
  int lane = r >> 3, i = r & 7;
  int bk = block / NT, bn = block % NT;
  int n = bn * 16 + (lane & 15);
  int k = bk * 32 + ((i >> 2) << 4) + ((lane >> 4) << 3) + ((i & 3) << 1);
  PackedHalf2 ph;
  ph.h[0] = (__bf16)W[(size_t)k * N + n];
  ph.h[1] = (__bf16)W[(size_t)(k + 1) * N + n];
  out[t] = ph.u;
}

// ---------------- WMMA GEMM: C[M x NT*16] = A[M x KT*32] * B ---------------
// One wave per 16x16 output tile; per-lane-contiguous packed blocks so each
// K-step is 2x global_load_b128 per operand + one v_wmma_f32_16x16x32_bf16.
template<int NT, int KT>
__global__ __launch_bounds__(256) void k_gemm_wmma(
    const uint32_t* __restrict__ Ap, const uint32_t* __restrict__ Bp,
    __bf16* __restrict__ C, int mTiles) {
  const int wave = (blockIdx.x * blockDim.x + threadIdx.x) >> 5;
  const int lane = threadIdx.x & 31;
  const int bm = wave / NT;
  const int bn = wave % NT;
  if (bm >= mTiles) return;

  v8f acc = {};
  const uint32_t* arow = Ap + (size_t)(bm * KT) * 256 + lane * 8;
  const uint32_t* bcol = Bp + (size_t)bn * 256 + lane * 8;
#pragma unroll
  for (int bk = 0; bk < KT; ++bk) {
    BlockHalf A, B;
    A.u[0] = *(const uint4*)(arow);
    A.u[1] = *(const uint4*)(arow + 4);
    B.u[0] = *(const uint4*)(bcol);
    B.u[1] = *(const uint4*)(bcol + 4);
    acc = __builtin_amdgcn_wmma_f32_16x16x32_bf16(
        false, A.v, false, B.v, (short)0, acc, false, false);
    arow += 256;
    bcol += NT * 256;
  }

  // D layout: lane L -> N = L&15, VGPR r -> M = (L>>4)*8 + r
  const int n  = bn * 16 + (lane & 15);
  const int m0 = bm * 16 + ((lane >> 4) << 3);
  const int ldc = NT * 16;
#pragma unroll
  for (int r = 0; r < 8; ++r)
    C[(size_t)(m0 + r) * ldc + n] = (__bf16)acc[r];
}

// ---------------- edge scatter: acc[dst] += norm * h[src] ------------------
// One wave per edge, FP = F/32 bf16 features per lane (vectorized b128/b64
// gather from the L2-resident bf16 feature matrix), f32 atomics to accum.
template<int F>
__global__ void k_scatter(const long long* __restrict__ src,
                          const long long* __restrict__ dst,
                          const __bf16* __restrict__ h,
                          const float* __restrict__ dinv,
                          float* __restrict__ acc, int E) {
  constexpr int FP = F / 32;
  long long gid = (long long)blockIdx.x * blockDim.x + threadIdx.x;
  int e = (int)(gid >> 5);
  if (e >= E) return;
  int lane = (int)(gid & 31);
  int s = (int)src[e];
  int d = (int)dst[e];
  float nrm = dinv[s] * dinv[d];

  union { uint32_t u[FP / 2]; __bf16 hv[FP]; } blk;
  const uint32_t* hp = (const uint32_t*)(h + (size_t)s * F) + lane * (FP / 2);
#pragma unroll
  for (int j = 0; j < FP / 2; ++j) blk.u[j] = hp[j];

  float* ap = acc + (size_t)d * F + lane * FP;
#pragma unroll
  for (int j = 0; j < FP; ++j)
    atomicAdd(&ap[j], nrm * (float)blk.hv[j]);
}

// ---------------- epilogues ------------------------------------------------
// layer1: acc = relu(acc + h*dinv^2 + bias)   (self-loop folded in)
__global__ void k_epilogue_relu(float* __restrict__ acc,
                                const __bf16* __restrict__ hs,
                                const float* __restrict__ dinv,
                                const float* __restrict__ bias,
                                int Nn, int F) {
  long long idx = (long long)blockIdx.x * blockDim.x + threadIdx.x;
  if (idx >= (long long)Nn * F) return;
  int i = (int)(idx / F);
  int f = (int)(idx - (long long)i * F);
  float di = dinv[i];
  float v = acc[idx] + (float)hs[idx] * di * di + bias[f];
  acc[idx] = v > 0.0f ? v : 0.0f;
}

// layer2: out += h*dinv^2 + bias
__global__ void k_epilogue_out(float* __restrict__ outp,
                               const __bf16* __restrict__ hs,
                               const float* __restrict__ dinv,
                               const float* __restrict__ bias,
                               int Nn, int F) {
  long long idx = (long long)blockIdx.x * blockDim.x + threadIdx.x;
  if (idx >= (long long)Nn * F) return;
  int i = (int)(idx / F);
  int f = (int)(idx - (long long)i * F);
  float di = dinv[i];
  outp[idx] += (float)hs[idx] * di * di + bias[f];
}

// ---------------------------------------------------------------------------

extern "C" void kernel_launch(void* const* d_in, const int* in_sizes, int n_in,
                              void* d_out, int out_size, void* d_ws, size_t ws_size,
                              hipStream_t stream) {
  (void)n_in; (void)out_size; (void)ws_size;

  const int   N    = 100000;
  const int   D_IN = 256, D_HID = 256, D_OUT = 128;
  const int   E    = in_sizes[1] / 2;

  const float*     x    = (const float*)d_in[0];
  const long long* eidx = (const long long*)d_in[1];     // int64 per reference
  const long long* src  = eidx;
  const long long* dst  = eidx + E;
  const float*     W1   = (const float*)d_in[2];
  const float*     b1   = (const float*)d_in[3];
  const float*     W2   = (const float*)d_in[4];
  const float*     b2   = (const float*)d_in[5];
  float*           outp = (float*)d_out;

  // workspace carve-up (256B aligned)
  char* ws = (char*)d_ws;
  size_t off = 0;
  auto carve = [&](size_t bytes) -> void* {
    void* p = ws + off;
    off += (bytes + 255) & ~(size_t)255;
    return p;
  };
  float*    dinv = (float*)   carve((size_t)N * 4);
  uint32_t* xb   = (uint32_t*)carve((size_t)N * D_IN * 2);       // packed A, L1
  uint32_t* W1b  = (uint32_t*)carve((size_t)D_IN * D_HID * 2);   // packed B, L1
  uint32_t* W2b  = (uint32_t*)carve((size_t)D_HID * D_OUT * 2);  // packed B, L2
  __bf16*   h1b  = (__bf16*)  carve((size_t)N * D_HID * 2);      // x@W1 (bf16)
  float*    acc1 = (float*)   carve((size_t)N * D_HID * 4);      // scatter accum / hr
  uint32_t* hrb  = (uint32_t*)carve((size_t)N * D_HID * 2);      // packed A, L2
  __bf16*   h2b  = (__bf16*)  carve((size_t)N * D_OUT * 2);      // hr@W2 (bf16)

  const int TB = 256;
  auto blocks = [](long long n, int tb) { return (int)((n + tb - 1) / tb); };

  // 1) symmetric normalization: dinv = rsqrt(in-degree incl. self-loop)
  k_init_deg<<<blocks(N, TB), TB, 0, stream>>>(dinv, N);
  k_accum_deg<<<blocks(E, TB), TB, 0, stream>>>(dst, dinv, E);
  k_finalize_dinv<<<blocks(N, TB), TB, 0, stream>>>(dinv, N);

  // 2) pack operands to bf16 in WMMA register order
  k_pack_a<<<blocks((long long)N * D_IN / 2, TB), TB, 0, stream>>>(x, xb, N, D_IN);
  k_pack_b<<<blocks((long long)D_IN * D_HID / 2, TB), TB, 0, stream>>>(W1, W1b, D_IN, D_HID);
  k_pack_b<<<blocks((long long)D_HID * D_OUT / 2, TB), TB, 0, stream>>>(W2, W2b, D_HID, D_OUT);

  // 3) h1 = x @ W1  (6250 M-tiles x 16 N-tiles = 100000 waves)
  {
    int tiles = (N / 16) * (D_HID / 16);
    k_gemm_wmma<D_HID / 16, D_IN / 32>
        <<<tiles / 8, TB, 0, stream>>>(xb, W1b, h1b, N / 16);
  }

  // 4) layer-1 aggregate: acc1 = scatter(norm * h1[src] -> dst)
  k_zero_f32<<<blocks((long long)N * D_HID, TB), TB, 0, stream>>>(acc1, (long long)N * D_HID);
  k_scatter<256><<<blocks((long long)E * 32, TB), TB, 0, stream>>>(src, dst, h1b, dinv, acc1, E);

  // 5) hr = relu(acc1 + h1*dinv^2 + b1), then repack to bf16 A layout
  k_epilogue_relu<<<blocks((long long)N * D_HID, TB), TB, 0, stream>>>(acc1, h1b, dinv, b1, N, D_HID);
  k_pack_a<<<blocks((long long)N * D_HID / 2, TB), TB, 0, stream>>>(acc1, hrb, N, D_HID);

  // 6) h2 = hr @ W2  (6250 x 8 = 50000 waves)
  {
    int tiles = (N / 16) * (D_OUT / 16);
    k_gemm_wmma<D_OUT / 16, D_HID / 32>
        <<<tiles / 8, TB, 0, stream>>>(hrb, W2b, h2b, N / 16);
  }

  // 7) layer-2 aggregate directly into d_out, then self-loop + bias
  k_zero_f32<<<blocks((long long)N * D_OUT, TB), TB, 0, stream>>>(outp, (long long)N * D_OUT);
  k_scatter<128><<<blocks((long long)E * 32, TB), TB, 0, stream>>>(src, dst, h2b, dinv, outp, E);
  k_epilogue_out<<<blocks((long long)N * D_OUT, TB), TB, 0, stream>>>(outp, h2b, dinv, b2, N, D_OUT);
}
